// Sopa_50345606644197
// MI455X (gfx1250) — compile-verified, matrix operand order checked by reading, not run on hardware
//
#include <hip/hip_runtime.h>
#include <hip/hip_fp16.h>
#include <math.h>

// ---------------------------------------------------------------------------
// SoPa (max-plus pattern RNN) for MI455X / gfx1250, wave32 + WMMA.
//
// Stage 1: convert x (16384x300 f32) and diags (4200x300 f32) to zero-padded
//          f16 (Dpad=320, Kpad=4224) in workspace.
// Stage 2: NT-GEMM tr = X * diags^T via v_wmma_f32_16x16x32_f16,
//          32x32 tile per wave, 64x128 per 256-thread block.
// Stage 3: sequential max-plus scan over L=512, one thread per (b,p),
//          bias added on the fly, tanh(score) streamed to out (L,B,P).
// ---------------------------------------------------------------------------

typedef __attribute__((ext_vector_type(16))) _Float16 v16h;
typedef __attribute__((ext_vector_type(8)))  _Float16 v8h;
typedef __attribute__((ext_vector_type(8)))  float    v8f;

#define L_SEQ   512
#define BATCH   32
#define DIM     300
#define DPAD    320            // 10 k-steps of 32
#define P_PAT   300
#define MLEN    7
#define KTOT    (2 * P_PAT * MLEN)   // 4200
#define KPAD    4224                 // 132 tiles of 32
#define NEGV    (-100.0f)

#define ROWS    (L_SEQ * BATCH)      // 16384

#define XH_BYTES  ((size_t)ROWS * DPAD * 2)   // 10,485,760
#define WH_BYTES  ((size_t)KPAD * DPAD * 2)   //  2,703,360

// ---------------------- Stage 1: f32 -> padded f16 -------------------------

__global__ void cvt_x_kernel(const float* __restrict__ x, _Float16* __restrict__ xh) {
  int i = blockIdx.x * blockDim.x + threadIdx.x;           // over ROWS*DPAD
  if (i >= ROWS * DPAD) return;
  int r = i / DPAD, d = i - r * DPAD;
  float v = (d < DIM) ? x[(size_t)r * DIM + d] : 0.0f;
  xh[i] = (_Float16)v;
}

__global__ void cvt_w_kernel(const float* __restrict__ w, _Float16* __restrict__ wh) {
  int i = blockIdx.x * blockDim.x + threadIdx.x;           // over KPAD*DPAD
  if (i >= KPAD * DPAD) return;
  int k = i / DPAD, d = i - k * DPAD;
  float v = (k < KTOT && d < DIM) ? w[(size_t)k * DIM + d] : 0.0f;
  wh[i] = (_Float16)v;
}

// ---------------------- Stage 2: WMMA GEMM ---------------------------------
// A/B 16-bit fragment layout (CDNA5 ISA, 16x32 f16):
//   lanes 0-15  (half=0): halves[0..7] = K 0..7,  halves[8..15] = K 16..23
//   lanes 16-31 (half=1): halves[0..7] = K 8..15, halves[8..15] = K 24..31
// Both A (row m) and B (col n) read contiguous d from row-major storage.

__device__ __forceinline__ v16h load_frag(const _Float16* __restrict__ row,
                                          int kk, int half) {
  const v8h lo = *reinterpret_cast<const v8h*>(row + kk + half * 8);       // 16B
  const v8h hi = *reinterpret_cast<const v8h*>(row + kk + 16 + half * 8);  // 16B
  v16h r;
#pragma unroll
  for (int i = 0; i < 8; ++i) { r[i] = lo[i]; r[i + 8] = hi[i]; }
  return r;
}

// C/D f32 16x16 layout: VGPR v, lane l: (M = v + 8*(l>>4), N = l&15)
__device__ __forceinline__ void store_tile(float* __restrict__ C, int rbase,
                                           int cbase, int m, int half, v8f acc) {
  int col = cbase + m;
  if (col >= KTOT) return;                       // K tail guard (4200..4223)
#pragma unroll
  for (int v = 0; v < 8; ++v) {
    int row = rbase + v + 8 * half;
    C[(size_t)row * KTOT + col] = acc[v];
  }
}

__global__ __launch_bounds__(256) void gemm_kernel(const _Float16* __restrict__ A,
                                                   const _Float16* __restrict__ Bm,
                                                   float* __restrict__ C) {
  const int lane = threadIdx.x & 31;
  const int wave = threadIdx.x >> 5;
  const int m    = lane & 15;
  const int half = lane >> 4;
  const int r0 = blockIdx.x * 64  + (wave >> 2) * 32;   // row tile base
  const int c0 = blockIdx.y * 128 + (wave & 3) * 32;    // col tile base

  const _Float16* a0p = A  + (size_t)(r0 + m)      * DPAD;
  const _Float16* a1p = A  + (size_t)(r0 + 16 + m) * DPAD;
  const _Float16* b0p = Bm + (size_t)(c0 + m)      * DPAD;
  const _Float16* b1p = Bm + (size_t)(c0 + 16 + m) * DPAD;

  v8f c00 = {}, c01 = {}, c10 = {}, c11 = {};

#pragma unroll 2
  for (int kk = 0; kk < DPAD; kk += 32) {
    v16h a0 = load_frag(a0p, kk, half);
    v16h a1 = load_frag(a1p, kk, half);
    v16h b0 = load_frag(b0p, kk, half);
    v16h b1 = load_frag(b1p, kk, half);
    c00 = __builtin_amdgcn_wmma_f32_16x16x32_f16(false, a0, false, b0,
                                                 (short)0, c00, false, false);
    c01 = __builtin_amdgcn_wmma_f32_16x16x32_f16(false, a0, false, b1,
                                                 (short)0, c01, false, false);
    c10 = __builtin_amdgcn_wmma_f32_16x16x32_f16(false, a1, false, b0,
                                                 (short)0, c10, false, false);
    c11 = __builtin_amdgcn_wmma_f32_16x16x32_f16(false, a1, false, b1,
                                                 (short)0, c11, false, false);
  }

  store_tile(C, r0,      c0,      m, half, c00);
  store_tile(C, r0,      c0 + 16, m, half, c01);
  store_tile(C, r0 + 16, c0,      m, half, c10);
  store_tile(C, r0 + 16, c0 + 16, m, half, c11);
}

// ---------------------- Stage 3: max-plus scan -----------------------------

__global__ __launch_bounds__(128) void sopa_scan_kernel(
    const float* __restrict__ tr, const float* __restrict__ bias,
    const float* __restrict__ epsilon, const long long* __restrict__ input_len,
    float* __restrict__ out) {
  int idx = blockIdx.x * blockDim.x + threadIdx.x;
  if (idx >= BATCH * P_PAT) return;
  int b = idx / P_PAT;
  int p = idx - b * P_PAT;
  int e = p / 50 + 1;                      // END_STATES[p]

  float h[MLEN];
  h[0] = 0.0f;
#pragma unroll
  for (int j = 1; j < MLEN; ++j) h[j] = NEGV;

  float eps[MLEN - 1];
#pragma unroll
  for (int j = 0; j < MLEN - 1; ++j) eps[j] = epsilon[p * (MLEN - 1) + j];

  float bs0[MLEN], bs1[MLEN];
#pragma unroll
  for (int j = 0; j < MLEN; ++j) {
    bs0[j] = bias[p * (2 * MLEN) + j];
    bs1[j] = bias[p * (2 * MLEN) + MLEN + j];
  }

  float s = NEGV;
  long long len = input_len[b];
  const float* trp = tr + (size_t)b * KTOT + (size_t)p * (2 * MLEN);
  float* outp = out + (size_t)b * P_PAT + p;

  for (int t = 0; t < L_SEQ; ++t) {
    const float* q = trp + (size_t)t * BATCH * KTOT;
    float t0[MLEN], t1[MLEN];
#pragma unroll
    for (int j = 0; j < MLEN; ++j) {
      t0[j] = q[j]        + bs0[j];        // self-loop scores
      t1[j] = q[MLEN + j] + bs1[j];        // advance scores
    }
    // epsilon transitions
    float ae[MLEN];
    ae[0] = fmaxf(h[0], NEGV);
#pragma unroll
    for (int j = 1; j < MLEN; ++j) ae[j] = fmaxf(h[j], h[j - 1] + eps[j - 1]);
    // advance (with restart at state 0) vs self-loop
    float nh[MLEN];
    nh[0] = fmaxf(0.0f, ae[0] + t0[0]);
#pragma unroll
    for (int j = 1; j < MLEN; ++j)
      nh[j] = fmaxf(ae[j - 1] + t1[j - 1], ae[j] + t0[j]);
#pragma unroll
    for (int j = 0; j < MLEN; ++j) h[j] = nh[j];

    float ev = h[1];
    ev = (e == 2) ? h[2] : ev;
    ev = (e == 3) ? h[3] : ev;
    ev = (e == 4) ? h[4] : ev;
    ev = (e == 5) ? h[5] : ev;
    ev = (e == 6) ? h[6] : ev;

    if (len >= (long long)t) s = fmaxf(s, ev);
    outp[(size_t)t * BATCH * P_PAT] = tanhf(s);
  }
}

// ---------------------- launch ---------------------------------------------

extern "C" void kernel_launch(void* const* d_in, const int* in_sizes, int n_in,
                              void* d_out, int out_size, void* d_ws, size_t ws_size,
                              hipStream_t stream) {
  (void)in_sizes; (void)n_in; (void)out_size; (void)ws_size;
  const float* x          = (const float*)d_in[0];       // (512,32,300)
  const float* diags      = (const float*)d_in[1];       // (4200,300)
  const float* bias       = (const float*)d_in[2];       // (4200,)
  const float* epsilon    = (const float*)d_in[3];       // (300,6)
  const long long* ilen   = (const long long*)d_in[4];   // (32,) int64
  float* out              = (float*)d_out;               // (512,32,300)

  char* ws = (char*)d_ws;
  _Float16* xh = (_Float16*)ws;                          // 10.5 MB
  _Float16* wh = (_Float16*)(ws + XH_BYTES);             //  2.7 MB
  float*    tr = (float*)(ws + XH_BYTES + WH_BYTES);     // 275.3 MB

  int nx = ROWS * DPAD;
  cvt_x_kernel<<<(nx + 255) / 256, 256, 0, stream>>>(x, xh);
  int nw = KPAD * DPAD;
  cvt_w_kernel<<<(nw + 255) / 256, 256, 0, stream>>>(diags, wh);

  dim3 gg(ROWS / 64, KPAD / 128);                        // 256 x 33
  gemm_kernel<<<gg, 256, 0, stream>>>(xh, wh, tr);

  sopa_scan_kernel<<<(BATCH * P_PAT + 127) / 128, 128, 0, stream>>>(
      tr, bias, epsilon, ilen, out);
}
